// EdgeGatedGraphConvolution_60232621359799
// MI455X (gfx1250) — matrix operand order repaired and checked by reading, not compiled, not run on hardware
//
#include <hip/hip_runtime.h>

#define HDIM 128
#define KZ   384          // 3*H
#define TM   16           // edge/node rows per tile
#define ZP   (KZ + 8)     // LDS pitch (ushorts) for z tile: shifts banks per row
#define AP   (HDIM + 8)   // LDS pitch for node A tile

typedef __attribute__((ext_vector_type(16))) __bf16 v16bf;
typedef __attribute__((ext_vector_type(8)))  float  v8f;

union ABFrag {
  v16bf v;
  uint4 q[2];
};

__device__ __forceinline__ unsigned short f2bf(float f) {
  unsigned int u = __float_as_uint(f);
  u += 0x7FFFu + ((u >> 16) & 1u);          // round-to-nearest-even
  return (unsigned short)(u >> 16);
}

__device__ __forceinline__ float sigmoidf(float x) {
  return 1.0f / (1.0f + __expf(-x));
}

// ---------------------------------------------------------------- utilities
__global__ void k_zero(float* __restrict__ p, long long n) {
  long long i = (long long)blockIdx.x * blockDim.x + threadIdx.x;
  const long long stride = (long long)gridDim.x * blockDim.x;
  for (; i < n; i += stride) p[i] = 0.0f;
}

__global__ void k_cvt_weights(const float* __restrict__ Wg, const float* __restrict__ Ws,
                              const float* __restrict__ Wd,
                              unsigned short* __restrict__ wgbf,
                              unsigned short* __restrict__ wsbf,
                              unsigned short* __restrict__ wdbf) {
  const int NG = HDIM * KZ, NS = HDIM * HDIM;
  int i = blockIdx.x * blockDim.x + threadIdx.x;
  if (i < NG)               wgbf[i] = f2bf(Wg[i]);
  else if (i < NG + NS)     wsbf[i - NG] = f2bf(Ws[i - NG]);
  else if (i < NG + 2 * NS) wdbf[i - NG - NS] = f2bf(Wd[i - NG - NS]);
}

// ------------------------------------------------- K1: edge GEMM (WMMA bf16)
// y[e, n] = sum_k z[e,k] * Wg[n,k] + bg[n]; z = [nf[src] | nf[dst] | ef]
// per-column sum/sumsq accumulated for BatchNorm phase 1.
__global__ __launch_bounds__(256) void k_edge_gemm(
    const long long* __restrict__ ei, const float* __restrict__ nf,
    const float* __restrict__ ef, const unsigned short* __restrict__ wgbf,
    const float* __restrict__ bg, float* __restrict__ y,
    float* __restrict__ zsum, float* __restrict__ zsumsq, int E)
{
  __shared__ __align__(16) unsigned short zt[TM * ZP];
  __shared__ int sidx[TM], didx[TM];

  const int t    = threadIdx.x;
  const int wave = t >> 5;        // 8 waves -> 8 column strips of 16
  const int lane = t & 31;
  const int mrow = lane & 15;
  const int kh   = lane >> 4;
  const int ncol = wave * 16 + mrow;

  // B fragments (row ncol of Wg in bf16), all 12 K-steps resident in VGPRs.
  // B layout (32x16 bf16): lane n = lane&15 holds K = kh*16 .. kh*16+15 contiguous.
  ABFrag bfr[12];
  #pragma unroll
  for (int ks = 0; ks < 12; ++ks) {
    const unsigned short* p = wgbf + ncol * KZ + ks * 32 + kh * 16;
    bfr[ks].q[0] = *(const uint4*)(p);
    bfr[ks].q[1] = *(const uint4*)(p + 8);
  }

  const float bias = bg[ncol];
  float lsum = 0.0f, lsq = 0.0f;
  const int ntiles = (E + TM - 1) / TM;

  for (int tile = blockIdx.x; tile < ntiles; tile += gridDim.x) {
    const int e0 = tile * TM;
    if (t < TM) {
      int ee = e0 + t; if (ee >= E) ee = E - 1;
      sidx[t] = (int)ei[ee];
      didx[t] = (int)ei[(long long)E + ee];
    }
    __syncthreads();

    // stage z tile [16 x 384] as bf16 (gathers hit L2-resident node features)
    for (int i = t; i < TM * KZ; i += 256) {
      const int m = i / KZ;
      const int k = i - m * KZ;
      int ee = e0 + m; if (ee >= E) ee = E - 1;
      float v;
      if (k < HDIM)          v = nf[(long long)sidx[m] * HDIM + k];
      else if (k < 2 * HDIM) v = nf[(long long)didx[m] * HDIM + (k - HDIM)];
      else                   v = ef[(long long)ee * HDIM + (k - 2 * HDIM)];
      zt[m * ZP + k] = f2bf(v);
    }
    __syncthreads();

    // A layout (16x32 bf16): lane holds row mrow; kh=0 -> K 0..15,
    // kh=1 -> K 16..31 contiguous (two 16-byte ds_load_b128 chunks).
    v8f c = {0.f, 0.f, 0.f, 0.f, 0.f, 0.f, 0.f, 0.f};
    #pragma unroll
    for (int ks = 0; ks < 12; ++ks) {
      const unsigned short* zp = &zt[mrow * ZP + ks * 32 + kh * 16];
      ABFrag a;
      a.q[0] = *(const uint4*)(zp);
      a.q[1] = *(const uint4*)(zp + 8);
      c = __builtin_amdgcn_wmma_f32_16x16x32_bf16(false, a.v, false, bfr[ks].v,
                                                  (short)0, c, false, false);
    }

    // C layout: VGPR r holds M = r + kh*8, column ncol
    float* yrow = y + (long long)e0 * HDIM + ncol;
    #pragma unroll
    for (int r = 0; r < 8; ++r) {
      const int m = r + kh * 8;
      if (e0 + m < E) {
        const float v = c[r] + bias;
        yrow[(long long)m * HDIM] = v;
        lsum += v;
        lsq  += v * v;
      }
    }
    __syncthreads();
  }

  atomicAdd(&zsum[ncol], lsum);
  atomicAdd(&zsumsq[ncol], lsq);
}

// ---------------------- K2: BN + SiLU in place; sigmoid scatter-add into agg
__global__ __launch_bounds__(256) void k_edge_bn_silu(
    const long long* __restrict__ ei, float* __restrict__ ue,
    const float* __restrict__ zsum, const float* __restrict__ zsumsq,
    const float* __restrict__ gamma, const float* __restrict__ beta,
    float* __restrict__ agg, int E)
{
  const long long idx = (long long)blockIdx.x * blockDim.x + threadIdx.x;
  const long long total = (long long)E * (HDIM / 4);
  if (idx >= total) return;
  const int e = (int)(idx >> 5);
  const int c = (int)(idx & 31) * 4;
  const int s = (int)ei[e];
  const float invE = 1.0f / (float)E;

  float4 v = *(float4*)(ue + (long long)e * HDIM + c);
  float4 r;
  #pragma unroll
  for (int j = 0; j < 4; ++j) {
    const float mean = zsum[c + j] * invE;
    const float var  = zsumsq[c + j] * invE - mean * mean;
    const float istd = rsqrtf(var + 1e-5f);
    const float bn   = ((&v.x)[j] - mean) * istd * gamma[c + j] + beta[c + j];
    const float act  = bn * sigmoidf(bn);           // SiLU
    (&r.x)[j] = act;
    atomicAdd(&agg[(long long)s * HDIM + c + j], sigmoidf(act));
  }
  *(float4*)(ue + (long long)e * HDIM + c) = r;
}

// -------------------------- K3: fused node GEMMs (Ws and Wd) via WMMA bf16
__global__ __launch_bounds__(256) void k_node_gemm(
    const float* __restrict__ nf,
    const unsigned short* __restrict__ wsbf, const unsigned short* __restrict__ wdbf,
    const float* __restrict__ bs, const float* __restrict__ bd,
    float* __restrict__ outS, float* __restrict__ outD, int N)
{
  __shared__ __align__(16) unsigned short at[TM * AP];
  const int t    = threadIdx.x;
  const int wave = t >> 5;
  const int lane = t & 31;
  const int mrow = lane & 15;
  const int kh   = lane >> 4;
  const int ncol = wave * 16 + mrow;

  ABFrag bS[4], bD[4];
  #pragma unroll
  for (int ks = 0; ks < 4; ++ks) {
    const unsigned short* ps = wsbf + ncol * HDIM + ks * 32 + kh * 16;
    bS[ks].q[0] = *(const uint4*)(ps);
    bS[ks].q[1] = *(const uint4*)(ps + 8);
    const unsigned short* pd = wdbf + ncol * HDIM + ks * 32 + kh * 16;
    bD[ks].q[0] = *(const uint4*)(pd);
    bD[ks].q[1] = *(const uint4*)(pd + 8);
  }
  const float biasS = bs[ncol], biasD = bd[ncol];
  const int ntiles = (N + TM - 1) / TM;

  for (int tile = blockIdx.x; tile < ntiles; tile += gridDim.x) {
    const int n0 = tile * TM;
    for (int i = t; i < TM * HDIM; i += 256) {
      const int m = i >> 7, k = i & 127;
      int rr = n0 + m; if (rr >= N) rr = N - 1;
      at[m * AP + k] = f2bf(nf[(long long)rr * HDIM + k]);
    }
    __syncthreads();

    v8f cs = {0.f, 0.f, 0.f, 0.f, 0.f, 0.f, 0.f, 0.f};
    v8f cd = {0.f, 0.f, 0.f, 0.f, 0.f, 0.f, 0.f, 0.f};
    #pragma unroll
    for (int ks = 0; ks < 4; ++ks) {
      const unsigned short* ap = &at[mrow * AP + ks * 32 + kh * 16];
      ABFrag a;
      a.q[0] = *(const uint4*)(ap);
      a.q[1] = *(const uint4*)(ap + 8);
      cs = __builtin_amdgcn_wmma_f32_16x16x32_bf16(false, a.v, false, bS[ks].v,
                                                   (short)0, cs, false, false);
      cd = __builtin_amdgcn_wmma_f32_16x16x32_bf16(false, a.v, false, bD[ks].v,
                                                   (short)0, cd, false, false);
    }

    float* rS = outS + (long long)n0 * HDIM + ncol;
    float* rD = outD + (long long)n0 * HDIM + ncol;
    #pragma unroll
    for (int r = 0; r < 8; ++r) {
      const int m = r + kh * 8;
      if (n0 + m < N) {
        rS[(long long)m * HDIM] = cs[r] + biasS;
        rD[(long long)m * HDIM] = cd[r] + biasD;
      }
    }
    __syncthreads();
  }
}

// ------- K4: normalized = sig/(agg[src]+eps); upd[src] += normalized * thee
__global__ __launch_bounds__(256) void k_edge_aggregate(
    const long long* __restrict__ ei, const float* __restrict__ ue,
    const float* __restrict__ agg, const float* __restrict__ nfWd,
    float* __restrict__ upd, int E)
{
  const long long idx = (long long)blockIdx.x * blockDim.x + threadIdx.x;
  const long long total = (long long)E * (HDIM / 4);
  if (idx >= total) return;
  const int e = (int)(idx >> 5);
  const int c = (int)(idx & 31) * 4;
  const int s = (int)ei[e];
  const int d = (int)ei[(long long)E + e];

  float4 u  = *(const float4*)(ue    + (long long)e * HDIM + c);
  float4 a  = *(const float4*)(agg   + (long long)s * HDIM + c);
  float4 th = *(const float4*)(nfWd  + (long long)d * HDIM + c);
  #pragma unroll
  for (int j = 0; j < 4; ++j) {
    const float sg  = sigmoidf((&u.x)[j]);
    const float nrm = sg / ((&a.x)[j] + 1e-6f);
    atomicAdd(&upd[(long long)s * HDIM + c + j], nrm * (&th.x)[j]);
  }
}

// -------------- K5a: h = nfWs + upd (in place into upd); column stats for BN
__global__ __launch_bounds__(128) void k_node_stats(
    const float* __restrict__ nfWs, float* __restrict__ hbuf,
    float* __restrict__ nsum, float* __restrict__ nsumsq, int N)
{
  const int c  = threadIdx.x;              // one thread per column, coalesced rows
  const int r0 = blockIdx.x * 64;
  int r1 = r0 + 64; if (r1 > N) r1 = N;
  float s = 0.0f, q = 0.0f;
  for (int r = r0; r < r1; ++r) {
    const long long i = (long long)r * HDIM + c;
    const float h = nfWs[i] + hbuf[i];
    hbuf[i] = h;
    s += h;
    q += h * h;
  }
  atomicAdd(&nsum[c], s);
  atomicAdd(&nsumsq[c], q);
}

// ---------------- K5b: updated_nodes = nf + SiLU(BN(h)) -> d_out node region
__global__ __launch_bounds__(256) void k_node_finish(
    const float* __restrict__ nf, const float* __restrict__ hbuf,
    const float* __restrict__ nsum, const float* __restrict__ nsumsq,
    const float* __restrict__ gamma, const float* __restrict__ beta,
    float* __restrict__ out, int N)
{
  const long long idx = (long long)blockIdx.x * blockDim.x + threadIdx.x;
  const long long total = (long long)N * (HDIM / 4);
  if (idx >= total) return;
  const int n = (int)(idx >> 5);
  const int c = (int)(idx & 31) * 4;
  const float invN = 1.0f / (float)N;

  float4 h = *(const float4*)(hbuf + (long long)n * HDIM + c);
  float4 x = *(const float4*)(nf   + (long long)n * HDIM + c);
  float4 r;
  #pragma unroll
  for (int j = 0; j < 4; ++j) {
    const float mean = nsum[c + j] * invN;
    const float var  = nsumsq[c + j] * invN - mean * mean;
    const float istd = rsqrtf(var + 1e-5f);
    const float bn   = ((&h.x)[j] - mean) * istd * gamma[c + j] + beta[c + j];
    (&r.x)[j] = (&x.x)[j] + bn * sigmoidf(bn);
  }
  *(float4*)(out + (long long)n * HDIM + c) = r;
}

// ---------------------------------------------------------------------------
extern "C" void kernel_launch(void* const* d_in, const int* in_sizes, int n_in,
                              void* d_out, int out_size, void* d_ws, size_t ws_size,
                              hipStream_t stream) {
  (void)n_in; (void)out_size; (void)ws_size;

  const long long* ei = (const long long*)d_in[0];   // edge_index int64 [2,E]
  const float* nf = (const float*)d_in[1];
  const float* ef = (const float*)d_in[2];
  const float* Wg = (const float*)d_in[3];
  const float* bg = (const float*)d_in[4];
  const float* Ws = (const float*)d_in[5];
  const float* bs = (const float*)d_in[6];
  const float* Wd = (const float*)d_in[7];
  const float* bd = (const float*)d_in[8];
  const float* zg = (const float*)d_in[9];
  const float* zb = (const float*)d_in[10];
  const float* ng = (const float*)d_in[11];
  const float* nb = (const float*)d_in[12];

  const int E = in_sizes[0] / 2;          // 640000
  const int N = in_sizes[1] / HDIM;       // 20000
  const long long NH = (long long)N * HDIM;

  float* out_nodes = (float*)d_out;       // [N,H]
  float* out_edges = out_nodes + NH;      // [E,H] (y staged here, BN'd in place)

  float* ws     = (float*)d_ws;
  float* zsum   = ws;
  float* zsumsq = ws + 128;
  float* nsum   = ws + 256;
  float* nsumsq = ws + 384;
  float* agg    = ws + 512;               // [N,H]
  float* upd    = agg + NH;               // [N,H] (becomes h)
  float* nfWs   = upd + NH;               // [N,H]
  float* nfWd   = nfWs + NH;              // [N,H]
  unsigned short* wgbf = (unsigned short*)(nfWd + NH);   // bf16 [128,384]
  unsigned short* wsbf = wgbf + HDIM * KZ;               // bf16 [128,128]
  unsigned short* wdbf = wsbf + HDIM * HDIM;             // bf16 [128,128]

  // zero stats + agg + upd (graph-replay safe: re-zeroed every call)
  k_zero<<<2048, 256, 0, stream>>>(ws, 512 + 2 * NH);

  // fp32 weights -> bf16 once per call
  {
    const int tot = HDIM * KZ + 2 * HDIM * HDIM;
    k_cvt_weights<<<(tot + 255) / 256, 256, 0, stream>>>(Wg, Ws, Wd, wgbf, wsbf, wdbf);
  }

  // K1: persistent WMMA edge GEMM + BN phase-1 stats
  k_edge_gemm<<<2048, 256, 0, stream>>>(ei, nf, ef, wgbf, bg, out_edges, zsum, zsumsq, E);

  const long long etotal = (long long)E * (HDIM / 4);
  const int eblocks = (int)((etotal + 255) / 256);

  // K2: BN + SiLU in place, sigmoid scatter-sum into agg
  k_edge_bn_silu<<<eblocks, 256, 0, stream>>>(ei, out_edges, zsum, zsumsq, zg, zb, agg, E);

  // K3: node GEMMs (Ws & Wd fused)
  {
    int g = (N + TM - 1) / TM; if (g > 2048) g = 2048;
    k_node_gemm<<<g, 256, 0, stream>>>(nf, wsbf, wdbf, bs, bd, nfWs, nfWd, N);
  }

  // K4: normalize edges and scatter-aggregate into upd
  k_edge_aggregate<<<eblocks, 256, 0, stream>>>(ei, out_edges, agg, nfWd, upd, E);

  // K5: node BN stats then finish
  k_node_stats<<<(N + 63) / 64, 128, 0, stream>>>(nfWs, upd, nsum, nsumsq, N);
  {
    const long long ntotal = (long long)N * (HDIM / 4);
    k_node_finish<<<(int)((ntotal + 255) / 256), 256, 0, stream>>>(
        nf, upd, nsum, nsumsq, ng, nb, out_nodes, N);
  }
}
